// BahdanauAttention_45749991637023
// MI455X (gfx1250) — matrix operand order repaired and checked by reading, not compiled, not run on hardware
//
#include <hip/hip_runtime.h>
#include <hip/hip_bf16.h>

typedef __attribute__((ext_vector_type(16))) __bf16 v16bf;
typedef __attribute__((ext_vector_type(8)))  float  v8f;

constexpr int kB = 32, kS = 2048, kE = 1024, kD = 1024;
constexpr float kNegInf = -1e9f;
constexpr int kMT = 4;                                   // M-tiles per block (64 s-rows)

// Dynamic LDS partition for scores_kernel.
constexpr unsigned kLdsABytes  = kMT * 32 * 32 * 16 * 2; // 4 mtiles * 32 kc * 32 lanes * 16 bf16 = 128 KB
constexpr unsigned kLdsDhOff   = kLdsABytes;             // 4 KB
constexpr unsigned kLdsVOff    = kLdsDhOff + 4096;       // 4 KB
constexpr unsigned kLdsRedOff  = kLdsVOff + 4096;        // 8 waves * 64 rows * 4 B = 2 KB
constexpr unsigned kLdsTotal   = kLdsRedOff + 8 * 64 * 4;

__device__ __forceinline__ unsigned short f2bf(float x) {
    unsigned int u = __float_as_uint(x);
    u += 0x7FFFu + ((u >> 16) & 1u);          // round to nearest even
    return (unsigned short)(u >> 16);
}

// Branch-free tanh using v_exp_f32 / v_rcp_f32 directly.
__device__ __forceinline__ float fast_tanh(float x) {
    float xc = fminf(fmaxf(x, -10.f), 10.f);
    float e  = __builtin_amdgcn_exp2f(2.885390082f * xc);  // exp(2*xc)
    return (e - 1.f) * __builtin_amdgcn_rcpf(e + 1.f);
}

// ---------------------------------------------------------------------------
// 1) Swizzle W_enc (f32 [D,E]) into WMMA B-fragment order, bf16:
//    Wb[((d_tile*32 + k_chunk)*32 + lane)*16 + j]
//    N=lane%16, h=lane/16: K=8h+j (j<8), K=16+8h+(j-8) (j>=8)   (B is K x N = E x D)
// ---------------------------------------------------------------------------
__global__ void prep_w_kernel(const float* __restrict__ W_enc,
                              unsigned short* __restrict__ Wb) {
    int tile    = blockIdx.x;        // 64 d_tiles * 32 k_chunks = 2048
    int d_tile  = tile >> 5;
    int k_chunk = tile & 31;
    int i = threadIdx.x;             // 512 elems per tile, 2 per thread
    for (int it = 0; it < 2; ++it, i += 256) {
        int lane = i >> 4;
        int j    = i & 15;
        int d = d_tile * 16 + (lane & 15);
        int e = k_chunk * 32 + ((j < 8) ? ((lane >> 4) * 8 + j) : (16 + (lane >> 4) * 8 + (j - 8)));
        Wb[(((d_tile * 32 + k_chunk) * 32) + lane) * 16 + j] = f2bf(W_enc[d * kE + e]);
    }
}

// ---------------------------------------------------------------------------
// 2) dh[b,d] = sum_e dec[b,e] * W_dec[d,e]   (tiny GEMM, W_dec read once)
// ---------------------------------------------------------------------------
__global__ void dh_kernel(const float* __restrict__ dec_hidden,
                          const float* __restrict__ W_dec,
                          float* __restrict__ dh) {
    __shared__ float lds_w[8 * 1024];              // 8 rows of W_dec
    int d0 = blockIdx.x * 8;
    for (int idx = threadIdx.x; idx < 8 * 1024; idx += 256)
        lds_w[idx] = W_dec[(size_t)(d0 + (idx >> 10)) * kD + (idx & 1023)];
    __syncthreads();
    int b  = threadIdx.x >> 3;                     // 32 batches
    int dl = threadIdx.x & 7;                      // 8 rows
    const float* dec = dec_hidden + (size_t)b * kD;
    const float* wr  = lds_w + dl * 1024;
    float acc = 0.f;
    #pragma unroll 4
    for (int e = 0; e < kD; ++e) acc += dec[e] * wr[e];
    dh[(size_t)b * kD + d0 + dl] = acc;
}

// ---------------------------------------------------------------------------
// 3) Fused energy/scores kernel: per (b, 64-row s-tile), 8 waves.
//    scores[b,s] = sum_d v[d] * tanh(dh[b,d] + sum_e W_enc[d,e]*enc[b,s,e])
//    Four M-tiles per wave reuse each B fragment 4x (quarters L2 B-traffic).
// ---------------------------------------------------------------------------
__global__ void scores_kernel(const float* __restrict__ enc,
                              const unsigned short* __restrict__ Wb,
                              const float* __restrict__ dh,
                              const float* __restrict__ v,
                              const int* __restrict__ mask,
                              float* __restrict__ scores) {
    extern __shared__ __align__(32) char smem[];
    unsigned short* lds_A  = (unsigned short*)smem;            // fragment-swizzled A
    float*          lds_dh = (float*)(smem + kLdsDhOff);
    float*          lds_v  = (float*)(smem + kLdsVOff);
    float*          red    = (float*)(smem + kLdsRedOff);      // [8 waves][64 rows]

    int s_tile = blockIdx.x, b = blockIdx.y;
    int tid = threadIdx.x;
    int s0  = s_tile * (kMT * 16);

    for (int i = tid; i < 1024; i += 256) {
        lds_dh[i] = dh[(size_t)b * kD + i];
        lds_v[i]  = v[i];
    }

    // Load 64 x 1024 enc tile (f32, coalesced), convert bf16, store in
    // A-fragment-swizzled LDS layout: [(mt*32 + kc)*32 + lane]*16 + j.
    const float* encb = enc + (size_t)(b * kS + s0) * kE;
    for (int idx = tid; idx < kMT * 16 * 1024; idx += 256) {
        int m = idx >> 10, e = idx & 1023;
        float x = encb[(size_t)m * kE + e];
        int mt = m >> 4, ml = m & 15;
        int kc = e >> 5, kk = e & 31;
        int lane, j;
        if      (kk <  8) { lane = ml;      j = kk;      }
        else if (kk < 16) { lane = ml + 16; j = kk - 8;  }
        else if (kk < 24) { lane = ml;      j = kk - 8;  }
        else              { lane = ml + 16; j = kk - 16; }
        lds_A[((mt * 32 + kc) * 32 + lane) * 16 + j] = f2bf(x);
    }
    __syncthreads();

    int wave = tid >> 5, lane = tid & 31;
    const v16bf* Av = reinterpret_cast<const v16bf*>(lds_A);
    const v16bf* Bv = reinterpret_cast<const v16bf*>(Wb);

    float rowpart[kMT][8];
    #pragma unroll
    for (int mt = 0; mt < kMT; ++mt)
        #pragma unroll
        for (int r = 0; r < 8; ++r) rowpart[mt][r] = 0.f;

    // Each wave: 8 N-tiles of 16 columns, K = 1024 in 32 steps of 32,
    // four M-tiles sharing each B fragment.
    for (int i = 0; i < 8; ++i) {
        int ntile = wave * 8 + i;

        if (i < 7)
            __builtin_prefetch(&Bv[(size_t)((ntile + 1) * 32) * 32 + lane], 0, 3);

        v8f acc[kMT];
        #pragma unroll
        for (int mt = 0; mt < kMT; ++mt) acc[mt] = (v8f){};

        #pragma unroll 2
        for (int kc = 0; kc < 32; ++kc) {
            v16bf bb = Bv[(size_t)(ntile * 32 + kc) * 32 + lane];
            #pragma unroll
            for (int mt = 0; mt < kMT; ++mt) {
                v16bf a = Av[(mt * 32 + kc) * 32 + lane];
                acc[mt] = __builtin_amdgcn_wmma_f32_16x16x32_bf16(
                    false, a, false, bb, (short)0, acc[mt], false, false);
            }
        }
        // Epilogue: energy -> tanh -> dot with v (d fixed per lane per tile)
        int d = ntile * 16 + (lane & 15);
        float dhv = lds_dh[d], vv = lds_v[d];
        #pragma unroll
        for (int mt = 0; mt < kMT; ++mt)
            #pragma unroll
            for (int r = 0; r < 8; ++r)
                rowpart[mt][r] += fast_tanh(acc[mt][r] + dhv) * vv;
    }

    // Reduce over 16 lanes of each half-wave (same M rows, different d).
    #pragma unroll
    for (int mt = 0; mt < kMT; ++mt) {
        #pragma unroll
        for (int r = 0; r < 8; ++r) {
            float x = rowpart[mt][r];
            x += __shfl_xor(x, 1);
            x += __shfl_xor(x, 2);
            x += __shfl_xor(x, 4);
            x += __shfl_xor(x, 8);
            rowpart[mt][r] = x;
        }
    }
    if ((lane & 15) == 0) {
        int mbase = (lane >> 4) * 8;   // lane 0 -> rows 0..7, lane 16 -> rows 8..15
        #pragma unroll
        for (int mt = 0; mt < kMT; ++mt)
            #pragma unroll
            for (int r = 0; r < 8; ++r)
                red[wave * 64 + mt * 16 + mbase + r] = rowpart[mt][r];
    }
    __syncthreads();

    if (tid < kMT * 16) {
        float s = 0.f;
        #pragma unroll
        for (int w = 0; w < 8; ++w) s += red[w * 64 + tid];
        int sidx = b * kS + s0 + tid;
        scores[sidx] = (mask[sidx] == 0) ? kNegInf : s;
    }
}

// ---------------------------------------------------------------------------
// 4) Softmax over S per batch row (exp computed once, normalized in place)
// ---------------------------------------------------------------------------
__global__ void softmax_kernel(const float* __restrict__ scores,
                               float* __restrict__ weights) {
    __shared__ float red[256];
    int b = blockIdx.x, tid = threadIdx.x;
    const float* sc = scores + (size_t)b * kS;
    float* wb = weights + (size_t)b * kS;

    float mx = -3.4e38f;
    for (int s = tid; s < kS; s += 256) mx = fmaxf(mx, sc[s]);
    red[tid] = mx; __syncthreads();
    for (int off = 128; off > 0; off >>= 1) {
        if (tid < off) red[tid] = fmaxf(red[tid], red[tid + off]);
        __syncthreads();
    }
    mx = red[0]; __syncthreads();

    float sum = 0.f;
    for (int s = tid; s < kS; s += 256) {
        float e = __expf(sc[s] - mx);
        wb[s] = e;                         // stash exponentials
        sum += e;
    }
    red[tid] = sum; __syncthreads();
    for (int off = 128; off > 0; off >>= 1) {
        if (tid < off) red[tid] += red[tid + off];
        __syncthreads();
    }
    float inv = 1.f / red[0];
    for (int s = tid; s < kS; s += 256) wb[s] *= inv;
}

// ---------------------------------------------------------------------------
// 5) context[b,e] = sum_s weights[b,s] * enc[b,s,e]
// ---------------------------------------------------------------------------
__global__ void zero_kernel(float* __restrict__ p, int n) {
    int i = blockIdx.x * 256 + threadIdx.x;
    if (i < n) p[i] = 0.f;
}

__global__ void context_kernel(const float* __restrict__ enc,
                               const float* __restrict__ weights,
                               float* __restrict__ ctx) {
    __shared__ float w[128];
    int sc = blockIdx.x, b = blockIdx.y, tid = threadIdx.x;
    int s0 = sc * 128;
    if (tid < 128) w[tid] = weights[(size_t)b * kS + s0 + tid];
    __syncthreads();
    float acc[4] = {0.f, 0.f, 0.f, 0.f};
    for (int s = 0; s < 128; ++s) {
        const float* row = enc + (size_t)(b * kS + s0 + s) * kE;
        float ws = w[s];
        #pragma unroll
        for (int j = 0; j < 4; ++j) acc[j] += ws * row[tid + j * 256];
    }
    #pragma unroll
    for (int j = 0; j < 4; ++j)
        atomicAdd(&ctx[(size_t)b * kE + tid + j * 256], acc[j]);
}

// ---------------------------------------------------------------------------
extern "C" void kernel_launch(void* const* d_in, const int* in_sizes, int n_in,
                              void* d_out, int out_size, void* d_ws, size_t ws_size,
                              hipStream_t stream) {
    (void)in_sizes; (void)n_in; (void)out_size; (void)ws_size;
    const float* dec_hidden = (const float*)d_in[0];
    const float* enc        = (const float*)d_in[1];
    const int*   mask       = (const int*)  d_in[2];
    const float* W_enc      = (const float*)d_in[3];
    const float* W_dec      = (const float*)d_in[4];
    const float* v          = (const float*)d_in[5];

    char* ws = (char*)d_ws;
    unsigned short* Wb = (unsigned short*)ws;                              // 2 MB
    float* dh     = (float*)(ws + (size_t)2 * 1024 * 1024);                // 128 KB
    float* scores = (float*)(ws + (size_t)2 * 1024 * 1024 + 128 * 1024);   // 256 KB

    float* ctx     = (float*)d_out;                 // [B, E]
    float* weights = (float*)d_out + kB * kE;       // [B, S]

    prep_w_kernel<<<2048, 256, 0, stream>>>(W_enc, Wb);
    dh_kernel<<<kD / 8, 256, 0, stream>>>(dec_hidden, W_dec, dh);
    scores_kernel<<<dim3(kS / (kMT * 16), kB), 256, kLdsTotal, stream>>>(enc, Wb, dh, v, mask, scores);
    softmax_kernel<<<kB, 256, 0, stream>>>(scores, weights);
    zero_kernel<<<(kB * kE + 255) / 256, 256, 0, stream>>>(ctx, kB * kE);
    context_kernel<<<dim3(kS / 128, kB), 256, 0, stream>>>(enc, weights, ctx);
}
